// Attention_Rel_Scl_42786464202794
// MI455X (gfx1250) — matrix operand order repaired and checked by reading, not compiled, not run on hardware
//
#include <hip/hip_runtime.h>
#include <hip/hip_bf16.h>
#include <math.h>

typedef __attribute__((ext_vector_type(16))) __bf16 v16bf;
typedef __attribute__((ext_vector_type(8)))  float  v8f;
typedef unsigned int u32x4 __attribute__((ext_vector_type(4)));
typedef int          i32x8 __attribute__((ext_vector_type(8)));
typedef int          i32x4 __attribute__((ext_vector_type(4)));

#define BB 4
#define SS 2048
#define EE 256
#define HH 8
#define DD 32
#define LN_EPS 1e-5f

// ---- WMMA fragment index maps (contiguous k = 16*(lane/16) + i, same for A and B, 16-bit, wave32) ----
__device__ __forceinline__ int kfA(int i, int lane) {
  int v = i >> 1, h = i & 1;
  int base = 2 * v;
  return base + ((lane >= 16) ? 16 : 0) + h;
}
__device__ __forceinline__ int kfB(int i, int lane) {
  int v = i >> 1, h = i & 1;
  return 2 * v + h + ((lane >= 16) ? 16 : 0);
}
// C/D 16x16 f32: row = r + (lane>=16 ? 8 : 0), col = lane&15

// ---- Tensor Data Mover: 2D bf16 tile load, D# per CDNA5 ISA ----
// 6-arg form: (u32x4 g0, i32x8 g1, i32x4, i32x4, i32x8, i32 cpol)
__device__ __forceinline__ void tdm_load_2d_bf16(const __bf16* gptr, __bf16* lptr,
                                                 unsigned tensor_d0, unsigned tensor_d1,
                                                 unsigned tile_d0, unsigned tile_d1,
                                                 unsigned long long stride0) {
  unsigned long long ga = (unsigned long long)(uintptr_t)gptr;
  unsigned la = (unsigned)(uintptr_t)lptr;          // LDS aperture low 32 = offset
  u32x4 g0;
  g0[0] = 1u;                                       // count=1 (valid), no gather
  g0[1] = la;                                       // lds_addr
  g0[2] = (unsigned)(ga & 0xffffffffull);           // global_addr[31:0]
  g0[3] = (unsigned)((ga >> 32) & 0x01ffffffull)    // global_addr[56:32]
        | (2u << 30);                               // type=2 ("image")
  i32x8 g1;
  g1[0] = (int)(1u << 16);                          // data_size=1 -> 2 bytes
  g1[1] = (int)((tensor_d0 & 0xffffu) << 16);                                   // dim0[15:0]
  g1[2] = (int)(((tensor_d0 >> 16) & 0xffffu) | ((tensor_d1 & 0xffffu) << 16)); // dim0[31:16], dim1[15:0]
  g1[3] = (int)(((tensor_d1 >> 16) & 0xffffu) | ((tile_d0 & 0xffffu) << 16));   // dim1[31:16], tile0
  g1[4] = (int)(tile_d1 & 0xffffu);                 // tile1 (tile2=0)
  g1[5] = (int)(stride0 & 0xffffffffull);           // dim0_stride[31:0]
  g1[6] = (int)((stride0 >> 32) & 0xffffull);       // dim0_stride[47:32]
  g1[7] = 0;
  i32x4 z4 = {0, 0, 0, 0};
  i32x8 z8 = {0, 0, 0, 0, 0, 0, 0, 0};
  __builtin_amdgcn_tensor_load_to_lds(g0, g1, z4, z4, z8, 0);
}

// =====================================================================
// Flash attention (emitted first so --disasm snippet shows the TDM path)
//   out = softmax(QK^T*s, mask)@V  +  Bias@V   (bias added post-softmax)
// one wave per (b,h,16-query tile); 32 keys/step; double-buffered
// tensor_load_to_lds pipelined with s_wait_tensorcnt
// =====================================================================
__global__ void flash_attn_kernel(const __bf16* __restrict__ qb,
                                  const __bf16* __restrict__ kb,
                                  const __bf16* __restrict__ vt,
                                  const int*    __restrict__ mask,
                                  const float*  __restrict__ bias_table,
                                  float*        __restrict__ ao) {
  __shared__ __bf16 ktile[2][32][DD];          // [buf][key][dim]   2x2KB
  __shared__ __bf16 vtile[2][DD][32];          // [buf][dim][key]   2x2KB
  __shared__ float  ptile[16][32];             // P: C-layout -> A-layout
  __shared__ float  btile[47];                 // bias diagonals for this step

  const int lane = threadIdx.x;
  const int ntiles = SS / 16;
  int bid = blockIdx.x;
  const int it = bid % ntiles;
  int tmp = bid / ntiles;
  const int h = tmp % HH;
  const int b = tmp / HH;
  const int i0 = it * 16;
  const size_t base   = ((size_t)b * HH + h) * SS;   // qb/kb [b,h,s,d]
  const size_t vtbase = ((size_t)b * HH + h) * DD;   // vt    [b,h,d,s]

  // Q tile: one 16x32 A-fragment (D==32), contiguous b128 loads
  v16bf qf;
  #pragma unroll
  for (int i = 0; i < 16; ++i)
    qf[i] = qb[(base + i0 + (lane & 15)) * DD + kfA(i, lane)];

  v8f acc0 = {}, acc1 = {}, bacc0 = {}, bacc1 = {};
  float m[8], lsum[8];
  #pragma unroll
  for (int r = 0; r < 8; ++r) { m[r] = -1e30f; lsum[r] = 0.0f; }

  // prologue: TDM tile 0 (K: 32 keys x 32 dims; V^T: 32 dims x 32 keys)
  tdm_load_2d_bf16(&kb[base * DD],   &ktile[0][0][0], DD, 32, DD, 32, DD);
  tdm_load_2d_bf16(&vt[vtbase * SS], &vtile[0][0][0], 32, DD, 32, DD, SS);

  const v8f zero = {};
  for (int j0 = 0; j0 < SS; j0 += 32) {
    const int buf = (j0 >> 5) & 1;
    if (j0 + 32 < SS) {                        // uniform: prefetch next tile pair
      tdm_load_2d_bf16(&kb[(base + j0 + 32) * DD], &ktile[buf ^ 1][0][0],
                       DD, 32, DD, 32, DD);
      tdm_load_2d_bf16(&vt[vtbase * SS + j0 + 32], &vtile[buf ^ 1][0][0],
                       32, DD, 32, DD, SS);
      __builtin_amdgcn_s_wait_tensorcnt(2);    // in-order: current pair resident
    } else {
      __builtin_amdgcn_s_wait_tensorcnt(0);
    }

    const int n   = lane & 15;
    const int j0n = j0 + n, j1n = j0 + 16 + n;

    // ---- scores: S = Q @ K^T (two 16x16 tiles), K frags from LDS ----
    v16bf kf0, kf1;
    #pragma unroll
    for (int i = 0; i < 16; ++i) {
      int kd = kfB(i, lane);                   // contiguous -> ds_load_b128
      kf0[i] = ktile[buf][n][kd];
      kf1[i] = ktile[buf][16 + n][kd];
    }
    v8f s0 = __builtin_amdgcn_wmma_f32_16x16x32_bf16(
        false, qf, false, kf0, (short)0, zero, false, false);
    v8f s1 = __builtin_amdgcn_wmma_f32_16x16x32_bf16(
        false, qf, false, kf1, (short)0, zero, false, false);

    const int mv0 = mask[b * SS + j0n];
    const int mv1 = mask[b * SS + j1n];

    // ---- stage bias diagonals: idx = i-j+S-1 spans 47 values per step ----
    {
      int idx0 = i0 - (j0 + 31) + (SS - 1);    // always in [0, 2S-2-46]
      btile[lane] = bias_table[(size_t)(idx0 + lane) * HH + h];
      if (lane < 15)
        btile[32 + lane] = bias_table[(size_t)(idx0 + 32 + lane) * HH + h];
    }

    // ---- online softmax over this 32-key step ----
    #pragma unroll
    for (int r = 0; r < 8; ++r) {
      float a = (mv0 == 0) ? -10000.0f : s0[r];
      float c = (mv1 == 0) ? -10000.0f : s1[r];
      float rmax = fmaxf(a, c);
      rmax = fmaxf(rmax, __shfl_xor(rmax, 1, 32));
      rmax = fmaxf(rmax, __shfl_xor(rmax, 2, 32));
      rmax = fmaxf(rmax, __shfl_xor(rmax, 4, 32));
      rmax = fmaxf(rmax, __shfl_xor(rmax, 8, 32));   // row max within 16-lane half
      float mnew = fmaxf(m[r], rmax);
      float corr = __expf(m[r] - mnew);
      float p0 = __expf(a - mnew);
      float p1 = __expf(c - mnew);
      float rs = p0 + p1;
      rs += __shfl_xor(rs, 1, 32); rs += __shfl_xor(rs, 2, 32);
      rs += __shfl_xor(rs, 4, 32); rs += __shfl_xor(rs, 8, 32);
      lsum[r] = lsum[r] * corr + rs;
      m[r] = mnew;
      acc0[r] *= corr; acc1[r] *= corr;
      int prow = r + ((lane >= 16) ? 8 : 0);
      ptile[prow][n]      = p0;
      ptile[prow][16 + n] = p1;
    }
    __syncthreads();

    // ---- P (LDS transpose), bias (LDS diagonals), V^T frags (LDS) ----
    v16bf pa, ba, vb0, vb1;
    #pragma unroll
    for (int i = 0; i < 16; ++i) {
      int ka = kfA(i, lane);
      pa[i] = (__bf16)ptile[lane & 15][ka];
      ba[i] = (__bf16)btile[(lane & 15) - ka + 31];  // bias[i0+row, j0+ka]
      int kd = kfB(i, lane);                   // contiguous -> ds_load_b128
      vb0[i] = vtile[buf][n][kd];              // dims 0..15
      vb1[i] = vtile[buf][16 + n][kd];         // dims 16..31
    }
    acc0 = __builtin_amdgcn_wmma_f32_16x16x32_bf16(
        false, pa, false, vb0, (short)0, acc0, false, false);
    acc1 = __builtin_amdgcn_wmma_f32_16x16x32_bf16(
        false, pa, false, vb1, (short)0, acc1, false, false);
    bacc0 = __builtin_amdgcn_wmma_f32_16x16x32_bf16(
        false, ba, false, vb0, (short)0, bacc0, false, false);
    bacc1 = __builtin_amdgcn_wmma_f32_16x16x32_bf16(
        false, ba, false, vb1, (short)0, bacc1, false, false);
    __syncthreads();
  }

  // ---- finalize: out = acc/l + bias_acc, write [B,S,E] fp32 ----
  #pragma unroll
  for (int r = 0; r < 8; ++r) {
    float inv = 1.0f / lsum[r];
    int srow = i0 + r + ((lane >= 16) ? 8 : 0);
    size_t o = ((size_t)b * SS + srow) * EE + h * DD;
    ao[o + (lane & 15)]      = acc0[r] * inv + bacc0[r];
    ao[o + 16 + (lane & 15)] = acc1[r] * inv + bacc1[r];
  }
}

// =====================================================================
// QKV projection  y = x @ W^T  (bf16 WMMA, fp32 accum); x staged in LDS.
// Q,K stored [b,h,s,d]; V stored transposed [b,h,d,s]
// =====================================================================
__global__ void qkv_proj_kernel(const float* __restrict__ x,
                                const float* __restrict__ Wq,
                                const float* __restrict__ Wk,
                                const float* __restrict__ Wv,
                                __bf16* __restrict__ qb,
                                __bf16* __restrict__ kb,
                                __bf16* __restrict__ vt) {
  __shared__ __bf16 xs[16][EE];                // 8 KB staging
  const int lane = threadIdx.x;
  const int mat  = blockIdx.y;                 // 0=Q 1=K 2=V
  const int rt   = blockIdx.x;
  const int b    = rt / (SS / 16);
  const int s0   = (rt % (SS / 16)) * 16;
  const float* W = (mat == 0) ? Wq : ((mat == 1) ? Wk : Wv);
  const float scale = (mat == 0) ? 0.0625f : 1.0f;   // 1/sqrt(E) folded at store

  // stage x tile -> bf16 LDS; each lane converts a contiguous 128-float run
  {
    const int r0 = lane >> 1;
    const int c0 = (lane & 1) * 128;
    const float* src = &x[((size_t)b * SS + (s0 + r0)) * EE + c0];
    #pragma unroll 8
    for (int t = 0; t < 128; t += 4) {
      float4 f = *(const float4*)(src + t);
      xs[r0][c0 + t + 0] = (__bf16)f.x;
      xs[r0][c0 + t + 1] = (__bf16)f.y;
      xs[r0][c0 + t + 2] = (__bf16)f.z;
      xs[r0][c0 + t + 3] = (__bf16)f.w;
    }
  }
  __syncthreads();

  const int row = lane & 15;
  #pragma unroll 1
  for (int nt = 0; nt < 16; ++nt) {            // 16 output col tiles
    v8f acc = {};
    #pragma unroll 2
    for (int c = 0; c < 8; ++c) {              // K = 256 in 8 chunks of 32
      v16bf af, wb;
      #pragma unroll
      for (int i = 0; i < 16; ++i) {
        af[i] = xs[row][kfA(i, lane) + 32 * c];          // two b128 ds loads
        int o  = nt * 16 + (lane & 15);
        wb[i] = (__bf16)W[(size_t)o * EE + kfB(i, lane) + 32 * c];
      }
      acc = __builtin_amdgcn_wmma_f32_16x16x32_bf16(
          false, af, false, wb, (short)0, acc, false, false);
    }
    // uniform branch hoisted out of the store loop -> clean store streams
    if (mat < 2) {
      __bf16* dst = (mat == 0) ? qb : kb;
      #pragma unroll
      for (int r = 0; r < 8; ++r) {
        int srow = s0 + r + ((lane >= 16) ? 8 : 0);
        int o    = nt * 16 + (lane & 15);
        int hh   = o >> 5, d = o & 31;
        dst[(((size_t)b * HH + hh) * SS + srow) * DD + d] = (__bf16)(acc[r] * scale);
      }
    } else {
      #pragma unroll
      for (int r = 0; r < 8; ++r) {
        int srow = s0 + r + ((lane >= 16) ? 8 : 0);
        int o    = nt * 16 + (lane & 15);
        int hh   = o >> 5, d = o & 31;
        vt[(((size_t)b * HH + hh) * DD + d) * SS + srow] = (__bf16)acc[r];
      }
    }
  }
}

// =====================================================================
// LayerNorm over E=256, one wave per (b,s) row
// =====================================================================
__global__ void layernorm_kernel(const float* __restrict__ ao,
                                 const float* __restrict__ gamma,
                                 const float* __restrict__ beta,
                                 float* __restrict__ out) {
  const int lane = threadIdx.x;
  const size_t rowoff = (size_t)blockIdx.x * EE;
  float v[8];
  float s = 0.0f;
  #pragma unroll
  for (int j = 0; j < 8; ++j) { v[j] = ao[rowoff + j * 32 + lane]; s += v[j]; }
  s += __shfl_xor(s, 1, 32);  s += __shfl_xor(s, 2, 32);
  s += __shfl_xor(s, 4, 32);  s += __shfl_xor(s, 8, 32);
  s += __shfl_xor(s, 16, 32);
  float mu = s * (1.0f / EE);
  float vs = 0.0f;
  #pragma unroll
  for (int j = 0; j < 8; ++j) { float d = v[j] - mu; vs += d * d; }
  vs += __shfl_xor(vs, 1, 32); vs += __shfl_xor(vs, 2, 32);
  vs += __shfl_xor(vs, 4, 32); vs += __shfl_xor(vs, 8, 32);
  vs += __shfl_xor(vs, 16, 32);
  float rstd = rsqrtf(vs * (1.0f / EE) + LN_EPS);
  #pragma unroll
  for (int j = 0; j < 8; ++j) {
    int c = j * 32 + lane;
    out[rowoff + c] = gamma[c] * (v[j] - mu) * rstd + beta[c];
  }
}

extern "C" void kernel_launch(void* const* d_in, const int* in_sizes, int n_in,
                              void* d_out, int out_size, void* d_ws, size_t ws_size,
                              hipStream_t stream) {
  const float* x     = (const float*)d_in[0];
  const int*   mask  = (const int*)  d_in[1];
  const float* Wq    = (const float*)d_in[2];
  const float* Wk    = (const float*)d_in[3];
  const float* Wv    = (const float*)d_in[4];
  const float* bt    = (const float*)d_in[5];
  const float* gamma = (const float*)d_in[6];
  const float* beta  = (const float*)d_in[7];
  float* out = (float*)d_out;

  const size_t N = (size_t)BB * SS * EE;       // 2,097,152 elements
  __bf16* qb   = (__bf16*)d_ws;                // 4 MB
  __bf16* kbuf = qb + N;                       // 4 MB
  __bf16* vtb  = kbuf + N;                     // 4 MB (transposed V)
  float*  ao   = (float*)(vtb + N);            // 8 MB (pre-LN fp32)

  qkv_proj_kernel<<<dim3(BB * SS / 16, 3), 32, 0, stream>>>(
      x, Wq, Wk, Wv, qb, kbuf, vtb);
  flash_attn_kernel<<<BB * HH * (SS / 16), 32, 0, stream>>>(
      qb, kbuf, vtb, mask, bt, ao);
  layernorm_kernel<<<BB * SS, 32, 0, stream>>>(ao, gamma, beta, out);
}
